// R_GCN_Layer_20822001451129
// MI455X (gfx1250) — compile-verified, matrix operand order checked by reading, not run on hardware
//
#include <hip/hip_runtime.h>

typedef __attribute__((ext_vector_type(16))) _Float16 v16h;
typedef __attribute__((ext_vector_type(8)))  float    v8f;

#define D     64
#define NREL  32
#define GRP   128           // binned edge slots handled per block (8 waves x 16 edges)

// workspace layout (ints)
#define CNT_OFF   0         // 32 counters
#define OFF_OFF   32        // 33 padded exclusive-scan offsets
#define CUR_OFF   66        // 32 scatter cursors
#define BINS_OFF  128       // binned edge ids, padded with -1

__global__ void zero_f32_kernel(float* p, long long n) {
  long long i = (long long)blockIdx.x * blockDim.x + threadIdx.x;
  if (i < n) p[i] = 0.0f;
}

__global__ void init_ws_kernel(int* ws, long long n) {
  long long i = (long long)blockIdx.x * blockDim.x + threadIdx.x;
  if (i < n) ws[i] = (i < BINS_OFF) ? 0 : -1;
}

__global__ void hist_kernel(const int* __restrict__ rel, int E, int* ws) {
  int e = blockIdx.x * blockDim.x + threadIdx.x;
  if (e < E) atomicAdd(&ws[CNT_OFF + rel[e]], 1);
}

__global__ void scan_kernel(int* ws) {
  if (blockIdx.x == 0 && threadIdx.x == 0) {
    int off = 0;
    for (int r = 0; r < NREL; ++r) {
      ws[OFF_OFF + r] = off;
      int c = ws[CNT_OFF + r];
      off += ((c + GRP - 1) / GRP) * GRP;   // pad each bin to GRP: a block never straddles bins
    }
    ws[OFF_OFF + NREL] = off;
  }
}

__global__ void scatter_kernel(const int* __restrict__ rel, int E, int* ws) {
  int e = blockIdx.x * blockDim.x + threadIdx.x;
  if (e < E) {
    int r = rel[e];
    int pos = ws[OFF_OFF + r] + atomicAdd(&ws[CUR_OFF + r], 1);
    ws[BINS_OFF + pos] = e;
  }
}

// Main pass: one block = 128 edges of a single relation; 8 waves x (16 edges -> 16x64 tile).
//
// CDNA5 16-bit A/B per-lane K pattern (ISA 7.12.2):
//   lane<16  : halves 0-7 hold K=klo..klo+7 with klo=0,  halves 8-15 hold K=16..23
//   lane>=16 : halves 0-7 hold K=8..15,                  halves 8-15 hold K=24..31
// i.e. element (K, n) of the B tile lives at
//   wfrag[K/32][n/16][(K%16>=8)*16 + n%16][(K&16 ? 8 : 0) + (K&7)]
__global__ __launch_bounds__(256) void rgcn_wmma_kernel(
    const float* __restrict__ h, const float* __restrict__ W,
    const int* __restrict__ src, const int* __restrict__ dst,
    const int* __restrict__ rel, const float* __restrict__ norm,
    float* __restrict__ out, const int* __restrict__ ws)
{
  __shared__ __align__(32) _Float16 wfrag[2][4][32][16]; // [kiter][ntile][lane][half]
  __shared__ int slot_e[GRP];
  __shared__ int sR;

  const int* bins = ws + BINS_OFF;
  int g   = blockIdx.x;
  int tid = threadIdx.x;

  if (tid == 0) sR = -1;
  __syncthreads();
  if (tid < GRP) {
    int e = bins[(long long)g * GRP + tid];
    slot_e[tid] = e;
    if (e >= 0) sR = rel[e];   // benign race: all valid edges in a group share one relation
  }
  __syncthreads();
  int r = sR;
  if (r < 0) return;           // fully padded group (uniform exit)

  // Stage W[r] (64x64 f32) into LDS in B-fragment layout, with coalesced float4 reads:
  // 4 vector loads per thread instead of 16 scalar loads.
  const float4* Wr4 = (const float4*)(W + (size_t)r * D * D);
  for (int qq = 0; qq < 4; ++qq) {
    int q  = tid + qq * 256;           // float4 index in [0, 1024)
    int K  = q >> 4;                   // row 0..63
    int n0 = (q & 15) << 2;            // first of 4 consecutive columns
    float4 f = Wr4[q];
    int kit   = K >> 5;
    int krem  = K & 31;
    int hi    = (krem & 15) >> 3;
    int t     = ((krem & 16) ? 8 : 0) + (krem & 7);
    int ntile = n0 >> 4;
    int lb    = hi * 16 + (n0 & 15);
    wfrag[kit][ntile][lb + 0][t] = (_Float16)f.x;
    wfrag[kit][ntile][lb + 1][t] = (_Float16)f.y;
    wfrag[kit][ntile][lb + 2][t] = (_Float16)f.z;
    wfrag[kit][ntile][lb + 3][t] = (_Float16)f.w;
  }
  __syncthreads();

  int lane = tid & 31;
  int wave = tid >> 5;
  int hi   = lane >> 4;
  int m    = lane & 15;

  int  e     = slot_e[wave * 16 + m];
  bool valid = (e >= 0);
  int   s    = valid ? src[e]  : 0;
  int   d    = valid ? dst[e]  : -1;           // sentinel for padded rows
  float nrm  = valid ? norm[e] : 0.0f;

  // A-fragments with norm folded in: (norm*h) @ W == norm * (h @ W), exact.
  v16h a[2];
  for (int kit = 0; kit < 2; ++kit) {
    v16h av;
    if (valid) {
      const float* hp = h + (size_t)s * D + kit * 32 + (hi ? 8 : 0);
      float4 f0 = ((const float4*)hp)[0];
      float4 f1 = ((const float4*)hp)[1];
      float4 f2 = ((const float4*)(hp + 16))[0];
      float4 f3 = ((const float4*)(hp + 16))[1];
      av[0]=(_Float16)(f0.x*nrm); av[1]=(_Float16)(f0.y*nrm); av[2]=(_Float16)(f0.z*nrm); av[3]=(_Float16)(f0.w*nrm);
      av[4]=(_Float16)(f1.x*nrm); av[5]=(_Float16)(f1.y*nrm); av[6]=(_Float16)(f1.z*nrm); av[7]=(_Float16)(f1.w*nrm);
      av[8]=(_Float16)(f2.x*nrm); av[9]=(_Float16)(f2.y*nrm); av[10]=(_Float16)(f2.z*nrm); av[11]=(_Float16)(f2.w*nrm);
      av[12]=(_Float16)(f3.x*nrm); av[13]=(_Float16)(f3.y*nrm); av[14]=(_Float16)(f3.z*nrm); av[15]=(_Float16)(f3.w*nrm);
    } else {
      for (int t = 0; t < 16; ++t) av[t] = (_Float16)0.0f;
    }
    a[kit] = av;
  }

  // Per-row destination pointers (D layout: VGPR j -> M = j + hi*8).
  // outp[j] = &out[dst_row * D + m]; column offset ntile*16 folds into the instr immediate.
  int allv = __all(valid);                     // wave-uniform fast path selector
  int dj[8]; float* outp[8];
  for (int j = 0; j < 8; ++j) {
    int row = hi * 8 + j;
    dj[j]   = __shfl(d, row, 32);
    outp[j] = out + (size_t)dj[j] * D + m;     // bogus if dj<0; never dereferenced then
  }

  // All 8 WMMAs back-to-back into 4 accumulators.
  v8f cacc[4];
  for (int ntile = 0; ntile < 4; ++ntile) {
    v8f c = {};
    for (int kit = 0; kit < 2; ++kit) {
      v16h b = *(const v16h*)&wfrag[kit][ntile][lane][0];
      c = __builtin_amdgcn_wmma_f32_16x16x32_f16(
              false, a[kit], false, b, (short)0, c, false, false);
    }
    cacc[ntile] = c;
  }

  if (allv) {
    // Fast path: no exec manipulation, immediate column offsets.
    for (int ntile = 0; ntile < 4; ++ntile)
      for (int j = 0; j < 8; ++j)
        atomicAdd(outp[j] + ntile * 16, cacc[ntile][j]);
  } else {
    for (int ntile = 0; ntile < 4; ++ntile)
      for (int j = 0; j < 8; ++j)
        if (dj[j] >= 0)
          atomicAdd(outp[j] + ntile * 16, cacc[ntile][j]);
  }
}

// Fallback if workspace is too small: scalar per-(edge,out) path, no workspace.
__global__ void rgcn_fallback_kernel(
    const float* __restrict__ h, const float* __restrict__ W,
    const int* __restrict__ src, const int* __restrict__ dst,
    const int* __restrict__ rel, const float* __restrict__ norm,
    float* __restrict__ out, int E)
{
  long long t = (long long)blockIdx.x * blockDim.x + threadIdx.x;
  if (t >= (long long)E * D) return;
  int e = (int)(t >> 6);
  int o = (int)(t & (D - 1));
  int s = src[e], rr = rel[e];
  const float* hp = h + (size_t)s * D;
  const float* wp = W + (size_t)rr * D * D + o;
  float acc = 0.0f;
  for (int i = 0; i < D; ++i) acc += hp[i] * wp[(size_t)i * D];
  atomicAdd(out + (size_t)dst[e] * D + o, acc * norm[e]);
}

extern "C" void kernel_launch(void* const* d_in, const int* in_sizes, int n_in,
                              void* d_out, int out_size, void* d_ws, size_t ws_size,
                              hipStream_t stream) {
  const float* h    = (const float*)d_in[0];
  const float* W    = (const float*)d_in[1];
  const int*   src  = (const int*)d_in[2];
  const int*   dst  = (const int*)d_in[3];
  const int*   rel  = (const int*)d_in[4];
  const float* norm = (const float*)d_in[5];
  float* out = (float*)d_out;
  int E = in_sizes[2];

  long long outN = (long long)out_size;
  zero_f32_kernel<<<(int)((outN + 255) / 256), 256, 0, stream>>>(out, outN);

  int binCap = ((E + GRP - 1) / GRP) * GRP + NREL * GRP;   // worst-case padded bin total
  size_t need = (size_t)(BINS_OFF + binCap) * sizeof(int);

  if (ws_size >= need) {
    int* ws = (int*)d_ws;
    long long initN = (long long)BINS_OFF + binCap;
    init_ws_kernel<<<(int)((initN + 255) / 256), 256, 0, stream>>>(ws, initN);
    hist_kernel<<<(E + 255) / 256, 256, 0, stream>>>(rel, E, ws);
    scan_kernel<<<1, 32, 0, stream>>>(ws);
    scatter_kernel<<<(E + 255) / 256, 256, 0, stream>>>(rel, E, ws);
    int groups = binCap / GRP;
    rgcn_wmma_kernel<<<groups, 256, 0, stream>>>(h, W, src, dst, rel, norm, out, ws);
  } else {
    long long tot = (long long)E * D;
    rgcn_fallback_kernel<<<(int)((tot + 255) / 256), 256, 0, stream>>>(h, W, src, dst, rel, norm, out, E);
  }
}
